// EquivariantBlock_34084860461595
// MI455X (gfx1250) — matrix operand order repaired and checked by reading, not compiled, not run on hardware
//
#include <hip/hip_runtime.h>
#include <math.h>

#define HDIM 128

typedef __attribute__((ext_vector_type(16))) __bf16 v16bf;
typedef __attribute__((ext_vector_type(8)))  float  v8f;

union FragU { v16bf v; __bf16 b[16]; uint4 q[2]; };

__device__ __forceinline__ float silu_f(float x) { return x / (1.0f + __expf(-x)); }

__device__ __forceinline__ void load8(float* d, const float* s) {
  float4 a = *(const float4*)s;
  float4 b = *(const float4*)(s + 4);
  d[0]=a.x; d[1]=a.y; d[2]=a.z; d[3]=a.w;
  d[4]=b.x; d[5]=b.y; d[6]=b.z; d[7]=b.w;
}

// B fragment (32x16 bf16): lane holds column n = nbase+(lane&15),
// elements are k = kbase + 8*(lane>>4) + i (i<8), kbase + 16 + 8*(lane>>4) + (i-8) (i>=8).
// Weights pre-transposed to [N][K] bf16, so this is two 16B loads.
__device__ __forceinline__ v16bf load_bT(const unsigned short* __restrict__ wT, int ldk,
                                         int kbase, int nbase, int lane) {
  int n  = nbase + (lane & 15);
  int kb = kbase + ((lane >> 4) << 3);
  const uint4* p = (const uint4*)(wT + (size_t)n * ldk + kb);
  FragU f;
  f.q[0] = p[0];
  f.q[1] = p[2];
  return f.v;
}

#define WMMA_BF16(a, b, c) \
  __builtin_amdgcn_wmma_f32_16x16x32_bf16(false, (a), false, (b), (short)0, (c), false, false)

// ---------------------------------------------------------------------------
// Weight prep: src f32 [K][128] row-major -> dst bf16 [128][K]
__global__ __launch_bounds__(256) void wprep(const float* __restrict__ src,
                                             unsigned short* __restrict__ dst, int K) {
  int idx = blockIdx.x * 256 + threadIdx.x;
  if (idx >= K * HDIM) return;
  int k = idx >> 7, n = idx & 127;
  union { __bf16 h; unsigned short s; } cv;
  cv.h = (__bf16)src[idx];
  dst[(size_t)n * K + k] = cv.s;
}

// ---------------------------------------------------------------------------
// radial + coord_diff precompute
__global__ __launch_bounds__(256) void prep_edges(const float* __restrict__ x,
                                                  const int* __restrict__ eidx, int E_,
                                                  float* __restrict__ radial,
                                                  float* __restrict__ cdiff) {
  int e = blockIdx.x * 256 + threadIdx.x;
  if (e >= E_) return;
  int r = eidx[e], c = eidx[E_ + e];
  float dx = x[r*3+0] - x[c*3+0];
  float dy = x[r*3+1] - x[c*3+1];
  float dz = x[r*3+2] - x[c*3+2];
  float rad = dx*dx + dy*dy + dz*dz;
  radial[e] = rad;
  float inv = 1.0f / (sqrtf(rad + 1e-8f) + 1.0f);
  cdiff[e*3+0] = dx * inv;
  cdiff[e*3+1] = dy * inv;
  cdiff[e*3+2] = dz * inv;
}

// ---------------------------------------------------------------------------
// Per-node precompute: outA = h @ Wa, outB = h @ Wb (Wa/Wb pre-transposed bf16)
__global__ __launch_bounds__(256) void node_linear(const float* __restrict__ h,
                                                   const unsigned short* __restrict__ wTa,
                                                   const unsigned short* __restrict__ wTb,
                                                   float* __restrict__ outA,
                                                   float* __restrict__ outB, int nTiles) {
  int tid = threadIdx.x, wid = tid >> 5, lane = tid & 31;
  int unit = blockIdx.x * 8 + wid;
  if (unit >= 2 * nTiles) return;
  int tile = unit >> 1, mat = unit & 1;
  const unsigned short* W = mat ? wTb : wTa;
  float* out = mat ? outB : outA;
  int half = lane >> 4, m = lane & 15;
  int node = tile * 16 + m;

  FragU afr[4];
#pragma unroll
  for (int kc = 0; kc < 4; ++kc) {
    int kb = kc * 32 + half * 8;
    float va[16];
    load8(va,     h + (size_t)node * HDIM + kb);
    load8(va + 8, h + (size_t)node * HDIM + kb + 16);
#pragma unroll
    for (int i = 0; i < 16; ++i) afr[kc].b[i] = (__bf16)va[i];
  }
  int f0 = lane & 15;
#pragma unroll
  for (int nt = 0; nt < 8; ++nt) {
    v8f c = {0.f,0.f,0.f,0.f,0.f,0.f,0.f,0.f};
#pragma unroll
    for (int kc = 0; kc < 4; ++kc)
      c = WMMA_BF16(afr[kc].v, load_bT(W, HDIM, kc * 32, nt * 16, lane), c);
    int f = nt * 16 + f0;
#pragma unroll
    for (int r = 0; r < 8; ++r)
      out[(size_t)(tile * 16 + r + 8 * half) * HDIM + f] = c[r];
  }
}

// ---------------------------------------------------------------------------
// Edge MLP + attention + scatter-add. Wave = 16 edges.
__global__ __launch_bounds__(256) void edge_gcl(
    const float* __restrict__ hA, const float* __restrict__ hB,
    const float* __restrict__ radial, const float* __restrict__ eattr,
    const int* __restrict__ eidx, int E_,
    const float* __restrict__ w1c, const float* __restrict__ b1,
    const unsigned short* __restrict__ w2T, const float* __restrict__ b2,
    const float* __restrict__ aw, const float* __restrict__ ab,
    float* __restrict__ agg, int eTiles) {
  int tid = threadIdx.x, wid = tid >> 5, lane = tid & 31;
  int tile = blockIdx.x * 8 + wid;
  if (tile >= eTiles) return;
  int half = lane >> 4, m = lane & 15;
  int eb = tile * 16;
  int e_m = eb + m;
  int r_i = eidx[e_m], c_i = eidx[E_ + e_m];
  float rad = radial[e_m], ea = eattr[e_m];
  float abv = ab[0];

  FragU afr[4];
#pragma unroll
  for (int kc = 0; kc < 4; ++kc) {
    int kb = kc * 32 + half * 8;
    float va[16], vb[16], w0[16], w1v[16], bb[16];
    load8(va,      hA + (size_t)r_i * HDIM + kb);
    load8(va + 8,  hA + (size_t)r_i * HDIM + kb + 16);
    load8(vb,      hB + (size_t)c_i * HDIM + kb);
    load8(vb + 8,  hB + (size_t)c_i * HDIM + kb + 16);
    load8(w0,      w1c + kb);
    load8(w0 + 8,  w1c + kb + 16);
    load8(w1v,     w1c + HDIM + kb);
    load8(w1v + 8, w1c + HDIM + kb + 16);
    load8(bb,      b1 + kb);
    load8(bb + 8,  b1 + kb + 16);
#pragma unroll
    for (int i = 0; i < 16; ++i) {
      float pre = va[i] + vb[i] + rad * w0[i] + ea * w1v[i] + bb[i];
      afr[kc].b[i] = (__bf16)silu_f(pre);
    }
  }

  int f0 = lane & 15;
  float b2v[8], awv[8];
#pragma unroll
  for (int nt = 0; nt < 8; ++nt) { b2v[nt] = b2[nt * 16 + f0]; awv[nt] = aw[nt * 16 + f0]; }

  float dot[8];
#pragma unroll
  for (int r = 0; r < 8; ++r) dot[r] = 0.f;
  float mval[8][8];
#pragma unroll
  for (int nt = 0; nt < 8; ++nt) {
    v8f c = {0.f,0.f,0.f,0.f,0.f,0.f,0.f,0.f};
#pragma unroll
    for (int kc = 0; kc < 4; ++kc)
      c = WMMA_BF16(afr[kc].v, load_bT(w2T, HDIM, kc * 32, nt * 16, lane), c);
#pragma unroll
    for (int r = 0; r < 8; ++r) {
      float mm = silu_f(c[r] + b2v[nt]);
      mval[nt][r] = mm;
      dot[r] += mm * awv[nt];
    }
  }
#pragma unroll
  for (int r = 0; r < 8; ++r) {
    float d = dot[r];
    d += __shfl_xor(d, 1, 32);
    d += __shfl_xor(d, 2, 32);
    d += __shfl_xor(d, 4, 32);
    d += __shfl_xor(d, 8, 32);
    dot[r] = 1.0f / (1.0f + __expf(-(d + abv)));   // sigmoid attention
  }
#pragma unroll
  for (int r = 0; r < 8; ++r) {
    int rr = __shfl(r_i, r + 8 * half, 32);        // row index of this edge
    float s = dot[r] * 0.01f;                      // / NORM_FACTOR
#pragma unroll
    for (int nt = 0; nt < 8; ++nt)
      atomicAdd(&agg[(size_t)rr * HDIM + nt * 16 + f0], mval[nt][r] * s);
  }
}

// ---------------------------------------------------------------------------
// Node model: h_out = h + (silu([h|agg]@W1 + b1) @ W2 + b2). Wave = 16 nodes.
__global__ __launch_bounds__(256) void node_update(
    const float* __restrict__ hin, const float* __restrict__ agg,
    const unsigned short* __restrict__ w1T, const float* __restrict__ b1,
    const unsigned short* __restrict__ w2T, const float* __restrict__ b2,
    float* __restrict__ hout, int nTiles) {
  __shared__ __bf16 tbuf[8][16 * HDIM];   // per-wave transpose scratch
  int tid = threadIdx.x, wid = tid >> 5, lane = tid & 31;
  int tile = blockIdx.x * 8 + wid;
  if (tile >= nTiles) return;
  int half = lane >> 4, m = lane & 15;
  int node = tile * 16 + m;

  v8f acc[8];
#pragma unroll
  for (int nt = 0; nt < 8; ++nt) acc[nt] = (v8f){0.f,0.f,0.f,0.f,0.f,0.f,0.f,0.f};

#pragma unroll
  for (int kc = 0; kc < 8; ++kc) {
    const float* src = (kc < 4) ? (hin + (size_t)node * HDIM) : (agg + (size_t)node * HDIM);
    int kb = (kc & 3) * 32 + half * 8;
    float va[16];
    load8(va,     src + kb);
    load8(va + 8, src + kb + 16);
    FragU a;
#pragma unroll
    for (int i = 0; i < 16; ++i) a.b[i] = (__bf16)va[i];
#pragma unroll
    for (int nt = 0; nt < 8; ++nt)
      acc[nt] = WMMA_BF16(a.v, load_bT(w1T, 2 * HDIM, kc * 32, nt * 16, lane), acc[nt]);
  }

  int f0 = lane & 15;
#pragma unroll
  for (int nt = 0; nt < 8; ++nt) {
    int f = nt * 16 + f0;
    float bb = b1[f];
#pragma unroll
    for (int r = 0; r < 8; ++r)
      tbuf[wid][(r + 8 * half) * HDIM + f] = (__bf16)silu_f(acc[nt][r] + bb);
  }
  asm volatile("" ::: "memory");   // same-wave DS ops are in-order; just stop reordering

  FragU a2[4];
#pragma unroll
  for (int kc = 0; kc < 4; ++kc) {
    const __bf16* base = &tbuf[wid][m * HDIM + kc * 32 + half * 8];
    a2[kc].q[0] = *(const uint4*)base;
    a2[kc].q[1] = *(const uint4*)(base + 16);
  }
#pragma unroll
  for (int nt = 0; nt < 8; ++nt) {
    v8f c = {0.f,0.f,0.f,0.f,0.f,0.f,0.f,0.f};
#pragma unroll
    for (int kc = 0; kc < 4; ++kc)
      c = WMMA_BF16(a2[kc].v, load_bT(w2T, HDIM, kc * 32, nt * 16, lane), c);
    int f = nt * 16 + f0;
    float bb = b2[f];
#pragma unroll
    for (int r = 0; r < 8; ++r) {
      size_t nd = (size_t)(tile * 16 + r + 8 * half) * HDIM + f;
      hout[nd] = hin[nd] + c[r] + bb;
    }
  }
}

// ---------------------------------------------------------------------------
// Equivariant coordinate update: cw = MLP(edge) @ c_w3 ; x += sum(cdiff*cw)/100
__global__ __launch_bounds__(256) void edge_coord(
    const float* __restrict__ hA, const float* __restrict__ hB,
    const float* __restrict__ radial, const float* __restrict__ eattr,
    const int* __restrict__ eidx, int E_,
    const float* __restrict__ w1c, const float* __restrict__ b1,
    const unsigned short* __restrict__ w2T, const float* __restrict__ b2,
    const float* __restrict__ w3,
    const float* __restrict__ cdiff, float* __restrict__ xout, int eTiles) {
  int tid = threadIdx.x, wid = tid >> 5, lane = tid & 31;
  int tile = blockIdx.x * 8 + wid;
  if (tile >= eTiles) return;
  int half = lane >> 4, m = lane & 15;
  int eb = tile * 16;
  int e_m = eb + m;
  int r_i = eidx[e_m], c_i = eidx[E_ + e_m];
  float rad = radial[e_m], ea = eattr[e_m];

  FragU afr[4];
#pragma unroll
  for (int kc = 0; kc < 4; ++kc) {
    int kb = kc * 32 + half * 8;
    float va[16], vb[16], w0[16], w1v[16], bb[16];
    load8(va,      hA + (size_t)r_i * HDIM + kb);
    load8(va + 8,  hA + (size_t)r_i * HDIM + kb + 16);
    load8(vb,      hB + (size_t)c_i * HDIM + kb);
    load8(vb + 8,  hB + (size_t)c_i * HDIM + kb + 16);
    load8(w0,      w1c + kb);
    load8(w0 + 8,  w1c + kb + 16);
    load8(w1v,     w1c + HDIM + kb);
    load8(w1v + 8, w1c + HDIM + kb + 16);
    load8(bb,      b1 + kb);
    load8(bb + 8,  b1 + kb + 16);
#pragma unroll
    for (int i = 0; i < 16; ++i) {
      float pre = va[i] + vb[i] + rad * w0[i] + ea * w1v[i] + bb[i];
      afr[kc].b[i] = (__bf16)silu_f(pre);
    }
  }

  int f0 = lane & 15;
  float dot[8];
#pragma unroll
  for (int r = 0; r < 8; ++r) dot[r] = 0.f;
#pragma unroll
  for (int nt = 0; nt < 8; ++nt) {
    v8f c = {0.f,0.f,0.f,0.f,0.f,0.f,0.f,0.f};
#pragma unroll
    for (int kc = 0; kc < 4; ++kc)
      c = WMMA_BF16(afr[kc].v, load_bT(w2T, HDIM, kc * 32, nt * 16, lane), c);
    int f = nt * 16 + f0;
    float bb = b2[f], ww = w3[f];
#pragma unroll
    for (int r = 0; r < 8; ++r) dot[r] += silu_f(c[r] + bb) * ww;
  }
#pragma unroll
  for (int r = 0; r < 8; ++r) {
    float d = dot[r];
    d += __shfl_xor(d, 1, 32);
    d += __shfl_xor(d, 2, 32);
    d += __shfl_xor(d, 4, 32);
    d += __shfl_xor(d, 8, 32);
    int em = eb + r + 8 * half;
    int rr = __shfl(r_i, r + 8 * half, 32);
    if (f0 < 3)
      atomicAdd(&xout[(size_t)rr * 3 + f0], cdiff[(size_t)em * 3 + f0] * d * 0.01f);
  }
}

// ---------------------------------------------------------------------------
extern "C" void kernel_launch(void* const* d_in, const int* in_sizes, int n_in,
                              void* d_out, int out_size, void* d_ws, size_t ws_size,
                              hipStream_t stream) {
  (void)n_in; (void)out_size; (void)ws_size;
  const float* h0    = (const float*)d_in[0];
  const float* x     = (const float*)d_in[1];
  const int*   eidx  = (const int*)d_in[2];
  const float* eattr = (const float*)d_in[3];
  const float* e_w1  = (const float*)d_in[4];
  const float* e_b1  = (const float*)d_in[5];
  const float* e_w2  = (const float*)d_in[6];
  const float* e_b2  = (const float*)d_in[7];
  const float* n_w1  = (const float*)d_in[8];
  const float* n_b1  = (const float*)d_in[9];
  const float* n_w2  = (const float*)d_in[10];
  const float* n_b2  = (const float*)d_in[11];
  const float* a_w   = (const float*)d_in[12];
  const float* a_b   = (const float*)d_in[13];
  const float* c_w1  = (const float*)d_in[14];
  const float* c_b1  = (const float*)d_in[15];
  const float* c_w2  = (const float*)d_in[16];
  const float* c_b2  = (const float*)d_in[17];
  const float* c_w3  = (const float*)d_in[18];

  const int N = in_sizes[0] / HDIM;
  const int E = in_sizes[3];
  const int nTiles = N / 16;
  const int eTiles = E / 16;

  float* hout = (float*)d_out;                       // [N,128]
  float* xout = hout + (size_t)N * HDIM;             // [N,3]

  float* hA     = (float*)d_ws;
  float* hB     = hA + (size_t)N * HDIM;
  float* agg    = hB + (size_t)N * HDIM;
  float* radial = agg + (size_t)N * HDIM;
  float* cdiff  = radial + E;
  unsigned short* wbf = (unsigned short*)(cdiff + (size_t)3 * E);

  const int M128 = HDIM * HDIM;          // 16384
  // bf16 transposed weights per layer: w1aT, w1bT, w2T, n_w1T(2*M128), n_w2T
  const size_t LSTRIDE = (size_t)M128 * 6;
  unsigned short* w1aT[2]; unsigned short* w1bT[2]; unsigned short* w2Tl[2];
  unsigned short* nw1T[2]; unsigned short* nw2T[2];
  for (int i = 0; i < 2; ++i) {
    unsigned short* base = wbf + i * LSTRIDE;
    w1aT[i] = base;
    w1bT[i] = base + M128;
    w2Tl[i] = base + 2 * M128;
    nw1T[i] = base + 3 * M128;     // [128][256]
    nw2T[i] = base + 5 * M128;
  }
  unsigned short* cw1aT = wbf + 2 * LSTRIDE;
  unsigned short* cw1bT = cw1aT + M128;
  unsigned short* cw2T  = cw1bT + M128;

  const int g128 = (M128 + 255) / 256, g256 = (2 * M128 + 255) / 256;
  for (int i = 0; i < 2; ++i) {
    const float* w1 = e_w1 + (size_t)i * 258 * HDIM;
    wprep<<<g128, 256, 0, stream>>>(w1,                          w1aT[i], HDIM);
    wprep<<<g128, 256, 0, stream>>>(w1 + M128,                   w1bT[i], HDIM);
    wprep<<<g128, 256, 0, stream>>>(e_w2 + (size_t)i * M128,     w2Tl[i], HDIM);
    wprep<<<g256, 256, 0, stream>>>(n_w1 + (size_t)i * 2 * M128, nw1T[i], 2 * HDIM);
    wprep<<<g128, 256, 0, stream>>>(n_w2 + (size_t)i * M128,     nw2T[i], HDIM);
  }
  wprep<<<g128, 256, 0, stream>>>(c_w1,        cw1aT, HDIM);
  wprep<<<g128, 256, 0, stream>>>(c_w1 + M128, cw1bT, HDIM);
  wprep<<<g128, 256, 0, stream>>>(c_w2,        cw2T,  HDIM);

  prep_edges<<<(E + 255) / 256, 256, 0, stream>>>(x, eidx, E, radial, cdiff);
  hipMemcpyAsync(xout, x, (size_t)N * 3 * sizeof(float), hipMemcpyDeviceToDevice, stream);

  for (int i = 0; i < 2; ++i) {
    const float* hin = (i == 0) ? h0 : hout;
    const float* w1  = e_w1 + (size_t)i * 258 * HDIM;
    node_linear<<<(2 * nTiles + 7) / 8, 256, 0, stream>>>(hin, w1aT[i], w1bT[i], hA, hB, nTiles);
    hipMemsetAsync(agg, 0, (size_t)N * HDIM * sizeof(float), stream);
    edge_gcl<<<(eTiles + 7) / 8, 256, 0, stream>>>(hA, hB, radial, eattr, eidx, E,
        w1 + 256 * HDIM, e_b1 + i * HDIM, w2Tl[i],
        e_b2 + i * HDIM, a_w + i * HDIM, a_b + i, agg, eTiles);
    node_update<<<(nTiles + 7) / 8, 256, 0, stream>>>(hin, agg,
        nw1T[i], n_b1 + i * HDIM, nw2T[i], n_b2 + i * HDIM, hout, nTiles);
  }

  node_linear<<<(2 * nTiles + 7) / 8, 256, 0, stream>>>(hout, cw1aT, cw1bT, hA, hB, nTiles);
  edge_coord<<<(eTiles + 7) / 8, 256, 0, stream>>>(hA, hB, radial, eattr, eidx, E,
      c_w1 + 256 * HDIM, c_b1, cw2T, c_b2, c_w3, cdiff, xout, eTiles);
}
